// MelNet_50869592654393
// MI455X (gfx1250) — matrix operand order repaired and checked by reading, not compiled, not run on hardware
//
#include <hip/hip_runtime.h>

// Problem constants (from reference): B=32, T=512, D=512, S=1024, M=16
#define B_   32
#define T_   512
#define D_   512
#define S_   1024
#define M_   16
#define KTOT 1536          // concat K: [x_t(512) | ctx(512) | h(512)]
#define NG   2048          // 4*D gate outputs
#define KT_TILES 48        // KTOT/32
#define NT_TILES 128       // NG/16
#define NBLK 32
#define TPB  256

typedef __attribute__((ext_vector_type(16))) __bf16   v16bf;
typedef __attribute__((ext_vector_type(8)))  float    v8f;
typedef __attribute__((ext_vector_type(8)))  unsigned v8u;
typedef __attribute__((ext_vector_type(4)))  unsigned v4u;
typedef __attribute__((ext_vector_type(4)))  unsigned su4;   // SGPR quad (D# group0)
typedef __attribute__((ext_vector_type(8)))  unsigned su8;   // SGPR octet (D# group1)

union Frag { v8u u8; v16bf bf; v4u u4[2]; };

__device__ __forceinline__ unsigned short f2bf(float f) {
  unsigned u = __float_as_uint(f);
  unsigned r = (u + 0x7fffu + ((u >> 16) & 1u)) >> 16;  // RNE
  return (unsigned short)r;
}
__device__ __forceinline__ float bf2f(unsigned short h) {
  return __uint_as_float(((unsigned)h) << 16);
}
__device__ __forceinline__ float sigf(float x)   { return 1.0f / (1.0f + __expf(-x)); }
__device__ __forceinline__ float tanhf_(float x) { return 2.0f * sigf(2.0f * x) - 1.0f; }

// ---------------- workspace layout (bytes) ----------------
#define OFF_WCAT  ((size_t)0)            // bf16 packed [48][128][32][8] dwords = 6291456 B
#define OFF_MEM   ((size_t)6291456)      // bf16 memory [32][1024][512]        = 33554432 B
#define OFF_XH    ((size_t)39845888)     // bf16 xh [32][1536]                 = 98304 B
#define OFF_GATES ((size_t)39944192)     // f32 gates [32][2048]               = 262144 B
#define OFF_H     ((size_t)40206336)     // f32 h [32][512]                    = 65536 B
#define OFF_C     ((size_t)40271872)     // f32 c [32][512]                    = 65536 B
#define OFF_KSI   ((size_t)40337408)     // f32 ksi [32][16]                   = 2048 B
#define OFF_BAR   ((size_t)40339456)     // 2 dwords barrier state

#define SMEM_BYTES (98304 + 4096 + 1024)

// ---------------- grid barrier (persistent kernel) ----------------
__device__ __forceinline__ void gsync(unsigned* cnt, unsigned* gen) {
  __threadfence();
  __syncthreads();
  if (threadIdx.x == 0) {
    volatile unsigned* vgen = (volatile unsigned*)gen;
    unsigned g = *vgen;
    if (atomicAdd(cnt, 1u) == (unsigned)(NBLK - 1)) {
      *cnt = 0u;
      __threadfence();
      *vgen = g + 1u;
    } else {
      while (*vgen == g) __builtin_amdgcn_s_sleep(2);
    }
  }
  __syncthreads();
  __threadfence();
}

// ---------------- TDM: one-shot 2D tile DMA  global -> LDS ----------------
// D# per CDNA5 ISA; 2-operand (<=2D) form, VADDR2/3 NULL.
// Tile = whole xh panel: 32 rows x 1536 bf16, row stride 1536 elements.
__device__ __forceinline__ void tdm_load_xh(const unsigned short* xh_g, unsigned lds_off) {
  unsigned long long ga = (unsigned long long)xh_g;
  su4 g0;
  g0.x = 1u;                                    // count=1 valid descriptor, user mode
  g0.y = lds_off;                               // lds_addr (bytes, wave LDS base relative)
  g0.z = (unsigned)(ga & 0xffffffffu);          // global_addr[31:0]
  g0.w = (unsigned)((ga >> 32) & 0x01ffffffu)   // global_addr[56:32]
         | (2u << 30);                          // type = 2 ("image")
  su8 g1;
  g1.s0 = (1u << 16);                           // workgroup_mask=0, data_size=1 (2 bytes)
  g1.s1 = ((unsigned)KTOT << 16);               // tensor_dim0[15:0] in bits[79:64->dw1 hi]
  g1.s2 = ((unsigned)B_ << 16);                 // tensor_dim0 hi=0 | tensor_dim1 lo = 32
  g1.s3 = ((unsigned)KTOT << 16);               // tensor_dim1 hi=0 | tile_dim0 = 1536
  g1.s4 = (unsigned)B_;                         // tile_dim1 = 32, tile_dim2 = 0
  g1.s5 = (unsigned)KTOT;                       // tensor_dim0_stride lo (elements)
  g1.s6 = 0u;                                   // stride hi | tensor_dim1_stride lo
  g1.s7 = 0u;
  asm volatile("tensor_load_to_lds %0, %1" :: "s"(g0), "s"(g1) : "memory");
}

// ---------------- one-time prep kernels ----------------
// Pack [W_ih | W_hh] into bf16, tiled for V_WMMA_F32_16X16X32_BF16 B (contiguous k order, same as A loads):
// tile (kt,nt): 32 lanes x 8 dwords contiguous; lane l -> N = nt*16 + (l&15),
// K group base = kt*32 + (l>>4)*16; dword v holds K pair (base+2v, base+2v+1).
__global__ void k_pack_w(const float* __restrict__ W_ih, const float* __restrict__ W_hh,
                         unsigned* __restrict__ wcat) {
  int idx = blockIdx.x * TPB + threadIdx.x;
  if (idx >= KT_TILES * NT_TILES * 32 * 8) return;
  int v    = idx & 7;
  int lane = (idx >> 3) & 31;
  int nt   = (idx >> 8) & (NT_TILES - 1);
  int kt   = idx >> 15;
  int n  = nt * 16 + (lane & 15);
  int k0 = kt * 32 + ((lane >> 4) << 4) + 2 * v;
  float f0, f1;
  if (k0 < 2 * D_) { f0 = W_ih[(size_t)n * (2 * D_) + k0];     f1 = W_ih[(size_t)n * (2 * D_) + k0 + 1]; }
  else             { f0 = W_hh[(size_t)n * D_ + (k0 - 2*D_)];  f1 = W_hh[(size_t)n * D_ + (k0 - 2*D_ + 1)]; }
  wcat[idx] = (unsigned)f2bf(f0) | ((unsigned)f2bf(f1) << 16);
}

__global__ void k_pack_mem(const float* __restrict__ mem, unsigned* __restrict__ out) {
  size_t idx = (size_t)blockIdx.x * TPB + threadIdx.x;
  if (idx >= (size_t)B_ * S_ * D_ / 2) return;
  float f0 = mem[2 * idx], f1 = mem[2 * idx + 1];
  out[idx] = (unsigned)f2bf(f0) | ((unsigned)f2bf(f1) << 16);
}

__global__ void k_init(const float* __restrict__ input, unsigned short* __restrict__ xh,
                       float* __restrict__ c, float* __restrict__ ksi, unsigned* __restrict__ bar) {
  int idx = blockIdx.x * TPB + threadIdx.x;
  if (idx < B_ * KTOT) {
    int b = idx / KTOT, k = idx % KTOT;
    xh[idx] = (k < D_) ? f2bf(input[((size_t)b * T_ + 0) * D_ + k]) : (unsigned short)0;
  }
  if (idx < B_ * D_) c[idx] = 0.f;
  if (idx < B_ * M_) ksi[idx] = 0.f;
  if (idx < 2) bar[idx] = 0u;
}

// ---------------- persistent step-loop kernel ----------------
__global__ __launch_bounds__(TPB) void k_persistent(
    const float* __restrict__ input, const float* __restrict__ b_ih,
    const float* __restrict__ b_hh, const float* __restrict__ W_g,
    const float* __restrict__ b_g, const unsigned* __restrict__ wcat,
    const unsigned* __restrict__ mem_bf, unsigned short* __restrict__ xh_g,
    float* __restrict__ gates, float* __restrict__ h_g, float* __restrict__ c_g,
    float* __restrict__ ksi_g, unsigned* __restrict__ bar,
    float* __restrict__ out_ctx, float* __restrict__ out_align)
{
  extern __shared__ char smem[];
  unsigned short* xh_lds = (unsigned short*)smem;               // [32][1536] bf16 (96KB)
  float* w_lds   = (float*)(smem + 98304);                      // [1024]
  float* phi_lds = (float*)(smem + 98304 + 4096);               // [48]
  float* ksi_lds = phi_lds + 48;                                // [16]
  float* rbe_lds = ksi_lds + 16;                                // [16] (1/beta)
  float* alp_lds = rbe_lds + 16;                                // [16]
  // dynamic LDS segment starts right after any static LDS
  const unsigned xh_lds_off = __builtin_amdgcn_groupstaticsize();

  unsigned* cnt = bar;
  unsigned* gen = bar + 1;
  const int tid  = threadIdx.x;
  const int lane = tid & 31;
  const int wv   = tid >> 5;
  const int b    = blockIdx.x;        // attention phase: one block per batch row

  // gates-phase tile mapping: 256 waves <-> 2x128 (16x16) output tiles
  const int gw = blockIdx.x * 8 + wv;
  const int mt = gw >> 7;
  const int nt = gw & (NT_TILES - 1);
  // A fragment addressing (contiguous k order, same as wcat packing): lanes 0-15 -> K base 0, 16-31 -> K base 16
  const int arow = lane & 15;
  const int koff = (lane >> 4) << 4;
  const unsigned* arow_dw = (const unsigned*)(xh_lds + (size_t)(mt * 16 + arow) * KTOT);
  const unsigned* bp_base = wcat + ((size_t)nt * 32 + lane) * 8;
  const int n_out  = nt * 16 + (lane & 15);
  const int m_base = mt * 16 + ((lane >> 4) << 3);
  const float gbias = b_ih[n_out] + b_hh[n_out];

  for (int t = 0; t < T_; ++t) {
    // -------- phase 0: TDM DMA of xh panel (bf16 [32][1536]) global -> LDS --------
    // One wave issues the tensor op (EXEC-independent, TENSORcnt-tracked), waits,
    // then the workgroup barrier releases the other 7 waves.
    if (wv == 0) {
      tdm_load_xh(xh_g, xh_lds_off);
      __builtin_amdgcn_s_wait_tensorcnt(0);
    }
    __syncthreads();

    // -------- phase 1: gates = xh @ [W_ih|W_hh]^T  (bf16 WMMA, f32 acc) --------
    {
      v8f acc = {0.f, 0.f, 0.f, 0.f, 0.f, 0.f, 0.f, 0.f};
      for (int kt = 0; kt < KT_TILES; ++kt) {
        Frag A, Bf;
        const unsigned* ap = arow_dw + kt * 16 + (koff >> 1);
        A.u4[0] = *(const v4u*)ap;            // K = koff .. koff+7   (ds_load_b128)
        A.u4[1] = *(const v4u*)(ap + 4);      // K = koff+8 .. koff+15
        Bf.u8 = *(const v8u*)(bp_base + (size_t)kt * (NT_TILES * 32 * 8));
        acc = __builtin_amdgcn_wmma_f32_16x16x32_bf16(
            false, A.bf, false, Bf.bf, (short)0, acc, false, false);
      }
      #pragma unroll
      for (int r = 0; r < 8; ++r)
        gates[(size_t)(m_base + r) * NG + n_out] = acc[r] + gbias;
    }
    gsync(cnt, gen);

    // -------- phase 2: LSTM elementwise; write h (f32 + bf16) and x_{t+1} --------
    for (int e = blockIdx.x * TPB + tid; e < B_ * D_; e += NBLK * TPB) {
      int bb = e >> 9, d = e & (D_ - 1);
      const float* gr = gates + (size_t)bb * NG;
      float ig = gr[d], fg = gr[D_ + d], gg = gr[2 * D_ + d], og = gr[3 * D_ + d];
      float cv = c_g[e];
      cv = sigf(fg) * cv + sigf(ig) * tanhf_(gg);
      float hv = sigf(og) * tanhf_(cv);
      c_g[e] = cv;
      h_g[e] = hv;
      xh_g[(size_t)bb * KTOT + 2 * D_ + d] = f2bf(hv);
      if (t + 1 < T_)
        xh_g[(size_t)bb * KTOT + d] = f2bf(input[((size_t)bb * T_ + (t + 1)) * D_ + d]);
    }
    gsync(cnt, gen);

    // -------- phase 3: GMM attention; block b owns batch row b --------
    // 3a: phi = h @ W_g^T + b_g  (48 dots of len 512; 8 waves x 6)
    for (int j = wv * 6; j < wv * 6 + 6; ++j) {
      float p = 0.f;
      const float* hr = h_g + (size_t)b * D_;
      const float* wr = W_g + (size_t)j * D_;
      for (int k = lane; k < D_; k += 32) p += hr[k] * wr[k];
      #pragma unroll
      for (int off = 16; off; off >>= 1) p += __shfl_down(p, off);
      if (lane == 0) phi_lds[j] = p + b_g[j];
    }
    __syncthreads();
    // 3b: ksi += exp(.); beta = exp(.); alpha = softmax(.)  (wave 0, lanes 0-15)
    if (wv == 0) {
      int m = lane & 15;
      bool lo = lane < 16;
      float kv = 0.f, bv = 1.f, av = -1e30f;
      if (lo) {
        kv = ksi_g[b * M_ + m] + __expf(phi_lds[m]);
        bv = __expf(phi_lds[M_ + m]);
        av = phi_lds[2 * M_ + m];
      }
      float mx = av;
      #pragma unroll
      for (int msk = 8; msk; msk >>= 1) mx = fmaxf(mx, __shfl_xor(mx, msk));
      float ex = __expf(av - mx);
      float sm = ex;
      #pragma unroll
      for (int msk = 8; msk; msk >>= 1) sm += __shfl_xor(sm, msk);
      if (lo) {
        ksi_g[b * M_ + m] = kv;
        ksi_lds[m] = kv;
        rbe_lds[m] = 1.0f / bv;
        alp_lds[m] = ex / sm;
      }
    }
    __syncthreads();
    // 3c: w[s] = sum_m alpha*(sig((s+1.5-ksi)/beta) - sig((s+0.5-ksi)/beta))
    for (int s = tid; s < S_; s += TPB) {
      float uR = (float)s + 1.5f, uL = (float)s + 0.5f;
      float acc = 0.f;
      #pragma unroll
      for (int m = 0; m < M_; ++m) {
        float rb = rbe_lds[m], kk = ksi_lds[m];
        acc += alp_lds[m] * (sigf((uR - kk) * rb) - sigf((uL - kk) * rb));
      }
      w_lds[s] = acc;
      out_align[((size_t)b * T_ + t) * S_ + s] = acc;
    }
    __syncthreads();
    // 3d: ctx = w @ memory[b]  (each thread: 2 adjacent d columns, bf16 dword loads)
    {
      int d0 = tid * 2;
      float c0 = 0.f, c1 = 0.f;
      const unsigned* mp = mem_bf + (size_t)b * S_ * (D_ / 2) + (d0 >> 1);
      for (int s = 0; s < S_; ++s) {
        unsigned pk = mp[(size_t)s * (D_ / 2)];
        float ws = w_lds[s];
        c0 += ws * bf2f((unsigned short)(pk & 0xffffu));
        c1 += ws * bf2f((unsigned short)(pk >> 16));
      }
      xh_g[(size_t)b * KTOT + D_ + d0]     = f2bf(c0);
      xh_g[(size_t)b * KTOT + D_ + d0 + 1] = f2bf(c1);
      size_t oi = ((size_t)b * T_ + t) * D_ + d0;
      out_ctx[oi]     = c0 + input[oi];
      out_ctx[oi + 1] = c1 + input[oi + 1];
    }
    gsync(cnt, gen);
  }
}

// ---------------- host-side launch ----------------
extern "C" void kernel_launch(void* const* d_in, const int* in_sizes, int n_in,
                              void* d_out, int out_size, void* d_ws, size_t ws_size,
                              hipStream_t stream) {
  const float* input  = (const float*)d_in[0];
  const float* memory = (const float*)d_in[1];
  const float* W_ih   = (const float*)d_in[2];
  const float* W_hh   = (const float*)d_in[3];
  const float* b_ih   = (const float*)d_in[4];
  const float* b_hh   = (const float*)d_in[5];
  const float* W_g    = (const float*)d_in[6];
  const float* b_g    = (const float*)d_in[7];

  char* ws = (char*)d_ws;
  unsigned*       wcat   = (unsigned*)(ws + OFF_WCAT);
  unsigned*       mem_bf = (unsigned*)(ws + OFF_MEM);
  unsigned short* xh     = (unsigned short*)(ws + OFF_XH);
  float*          gates  = (float*)(ws + OFF_GATES);
  float*          h_g    = (float*)(ws + OFF_H);
  float*          c_g    = (float*)(ws + OFF_C);
  float*          ksi    = (float*)(ws + OFF_KSI);
  unsigned*       bar    = (unsigned*)(ws + OFF_BAR);

  float* out_ctx   = (float*)d_out;
  float* out_align = out_ctx + (size_t)B_ * T_ * D_;

  // one-time (per call) packing: weights + memory to bf16, state init
  k_pack_w<<<(KT_TILES * NT_TILES * 32 * 8 + TPB - 1) / TPB, TPB, 0, stream>>>(W_ih, W_hh, wcat);
  k_pack_mem<<<(int)(((size_t)B_ * S_ * D_ / 2 + TPB - 1) / TPB), TPB, 0, stream>>>(memory, mem_bf);
  k_init<<<(B_ * KTOT + TPB - 1) / TPB, TPB, 0, stream>>>(input, xh, c_g, ksi, bar);

  // persistent sequential loop: 32 blocks (== B), 8 waves each
  k_persistent<<<NBLK, TPB, SMEM_BYTES, stream>>>(
      input, b_ih, b_hh, W_g, b_g, wcat, mem_bf, xh,
      gates, h_g, c_g, ksi, bar, out_ctx, out_align);
}
